// AttentionBlock_13572096656150
// MI455X (gfx1250) — matrix operand order repaired and checked by reading, not compiled, not run on hardware
//
#include <hip/hip_runtime.h>

typedef __attribute__((ext_vector_type(16))) __bf16 v16bf;
typedef __attribute__((ext_vector_type(8)))  float  v8f;

#define B_   32
#define C_   256
#define N_   1024
#define H_   8
#define HID_ 512
#define O3_  1536

// Padded LDS row strides (bf16 units). Multiple of 8 bf16 (=16B) keeps every
// per-lane fragment run 16-byte aligned for ds_load_b128; the +8/+8 pad breaks
// the power-of-two bank aliasing.
#define LD32 40   // panels with K extent 32 (uint stride 20)
#define LD64 72   // panels with K extent 64 (uint stride 36)

// ---------------- WMMA fragment helpers (CDNA5 16x16x32 bf16 layouts) -------

// A: 16(M) x 32(K) bf16, row-major in LDS with leading dim `ld`.
// lane L: m = L%16 ; element e -> k = ((e>>3)<<4) + ((L>>4)<<3) + (e&7)
// => two contiguous 16B runs per lane (vectorizes to 2x ds_load_b128).
__device__ inline v16bf frag_a(const __bf16* s, int lane, int ld) {
  const int m = lane & 15;
  const int koff = (lane >> 4) << 3;
  v16bf a;
#pragma unroll
  for (int e = 0; e < 16; ++e) {
    const int k = ((e >> 3) << 4) + koff + (e & 7);
    a[e] = s[m * ld + k];
  }
  return a;
}

// B: logical B[k][n], stored TRANSPOSED in LDS as tile[n][k] with k-stride ld.
// lane L: n = L%16 ; element e -> k = ((e>>3)<<4) + ((L>>4)<<3) + (e&7)
// => two contiguous 16B runs per lane (vectorizes to 2x ds_load_b128).
__device__ inline v16bf frag_bT(const __bf16* s, int lane, int ld) {
  const int n = lane & 15;
  const int kb = (lane >> 4) << 3;
  v16bf b;
#pragma unroll
  for (int e = 0; e < 16; ++e) b[e] = s[n * ld + ((e >> 3) << 4) + kb + (e & 7)];
  return b;
}

__device__ inline v8f wmma_bf16(v16bf a, v16bf b, v8f c) {
  return __builtin_amdgcn_wmma_f32_16x16x32_bf16(
      /*neg_a=*/false, a, /*neg_b=*/false, b,
      /*c_mod=*/(short)0, c, /*reuse_a=*/false, /*reuse_b=*/false);
}

// 2x2 bf16 register transpose helpers: u0 = {lo: x[k][n], hi: x[k][n+1]},
// u1 = {lo: x[k+1][n], hi: x[k+1][n+1]}  ->  column-packed pairs.
__device__ inline unsigned pack_lo(unsigned u0, unsigned u1) {
  return (u0 & 0xffffu) | (u1 << 16);          // {x[k][n], x[k+1][n]}
}
__device__ inline unsigned pack_hi(unsigned u0, unsigned u1) {
  return (u0 >> 16) | (u1 & 0xffff0000u);      // {x[k][n+1], x[k+1][n+1]}
}

// ---------------- precision conversion --------------------------------------

__global__ void cvt_f32_bf16(const float* __restrict__ in,
                             __bf16* __restrict__ out, int n) {
  int i = blockIdx.x * blockDim.x + threadIdx.x;
  if (i < n) out[i] = (__bf16)in[i];
}

// ---------------- QKV GEMM: qkv[b][o][n] = w_qkv[o][c] * x[b][c][n] ---------
// block = 256 threads = 8 waves; 64x64 output tile; K stepped 32.

__global__ __launch_bounds__(256) void gemm_qkv(const __bf16* __restrict__ wq,
                                                const __bf16* __restrict__ xb,
                                                __bf16* __restrict__ qkv) {
  __shared__ unsigned int As_u[64 * (LD32 / 2)];   // A: 64(o) x 32(c), padded
  __shared__ unsigned int Bt_u[64 * (LD32 / 2)];   // B^T: 64(n) x 32(c), padded
  const __bf16* As = (const __bf16*)As_u;
  const __bf16* Bt = (const __bf16*)Bt_u;

  const int tid = threadIdx.x, lane = tid & 31, wave = tid >> 5;
  const int mt = wave & 3, nt0 = (wave >> 2) << 1;
  const int n0 = blockIdx.x * 64;
  const int o0 = blockIdx.y * 64;
  const int b  = blockIdx.z;

  const unsigned int* wq_u = (const unsigned int*)wq;
  const unsigned int* xb_u = (const unsigned int*)(xb + (size_t)b * C_ * N_);

  v8f acc0 = {}, acc1 = {};
  for (int c0 = 0; c0 < C_; c0 += 32) {
    __syncthreads();
    // A panel: straight copy (row-major), 4 dwords/thread.
#pragma unroll
    for (int j = 0; j < 4; ++j) {
      int i = tid + j * 256;
      int r = i >> 4, cp = i & 15;
      As_u[r * (LD32 / 2) + cp] = wq_u[(o0 + r) * (C_ / 2) + (c0 >> 1) + cp];
    }
    // B panel: transpose-on-stage. 2 coalesced loads + 2 packed stores/thread.
#pragma unroll
    for (int t = 0; t < 2; ++t) {
      int i = tid + t * 256;           // 0..511
      int cp = i & 31;                 // n-pair
      int r  = i >> 5;                 // c-pair (0..15)
      unsigned u0 = xb_u[(c0 + 2 * r)     * (N_ / 2) + (n0 >> 1) + cp];
      unsigned u1 = xb_u[(c0 + 2 * r + 1) * (N_ / 2) + (n0 >> 1) + cp];
      Bt_u[(2 * cp)     * (LD32 / 2) + r] = pack_lo(u0, u1);
      Bt_u[(2 * cp + 1) * (LD32 / 2) + r] = pack_hi(u0, u1);
    }
    if (c0 + 32 < C_) {
      __builtin_prefetch(&wq_u[(o0 + (tid >> 4)) * (C_ / 2) + ((c0 + 32) >> 1) + (tid & 15)], 0, 1);
      __builtin_prefetch(&xb_u[(c0 + 32 + ((tid >> 5) << 1)) * (N_ / 2) + (n0 >> 1) + (tid & 31)], 0, 1);
    }
    __syncthreads();
    v16bf a  = frag_a(As + mt * 16 * LD32, lane, LD32);
    v16bf b0 = frag_bT(Bt + nt0 * 16 * LD32, lane, LD32);
    v16bf b1 = frag_bT(Bt + (nt0 + 1) * 16 * LD32, lane, LD32);
    acc0 = wmma_bf16(a, b0, acc0);
    acc1 = wmma_bf16(a, b1, acc1);
  }

  __bf16* outp = qkv + (size_t)b * O3_ * N_;
  const int col = lane & 15, rb = (lane >> 4) << 3;
#pragma unroll
  for (int r = 0; r < 8; ++r) {
    int row = o0 + mt * 16 + rb + r;
    outp[(size_t)row * N_ + n0 + nt0 * 16 + col]       = (__bf16)acc0[r];
    outp[(size_t)row * N_ + n0 + (nt0 + 1) * 16 + col] = (__bf16)acc1[r];
  }
}

// ---------------- softmax over d (q) + fold in SCALE, in place --------------

__global__ __launch_bounds__(128) void softmax_q(__bf16* __restrict__ qkv) {
  const int n = blockIdx.x * 128 + threadIdx.x;
  const int h = blockIdx.y;
  const int b = blockIdx.z;
  __bf16* base = qkv + ((size_t)b * O3_ + h * 64) * N_ + n;
  float v[64];
  float m = -1e30f;
#pragma unroll
  for (int d = 0; d < 64; ++d) {
    v[d] = (float)base[(size_t)d * N_];
    m = fmaxf(m, v[d]);
  }
  float s = 0.f;
#pragma unroll
  for (int d = 0; d < 64; ++d) { v[d] = __expf(v[d] - m); s += v[d]; }
  const float inv = 0.125f / s;   // SCALE = 64^-0.5 = 0.125
#pragma unroll
  for (int d = 0; d < 64; ++d) base[(size_t)d * N_] = (__bf16)(v[d] * inv);
}

// ---------------- softmax over n (k), in place ------------------------------

__global__ __launch_bounds__(256) void softmax_k(__bf16* __restrict__ qkv) {
  __shared__ float red[256];
  const int d = blockIdx.x, h = blockIdx.y, b = blockIdx.z;
  __bf16* row = qkv + ((size_t)b * O3_ + HID_ + h * 64 + d) * N_;
  const int tid = threadIdx.x;
  float v[4];
  float m = -1e30f;
#pragma unroll
  for (int j = 0; j < 4; ++j) { v[j] = (float)row[tid + j * 256]; m = fmaxf(m, v[j]); }
  red[tid] = m; __syncthreads();
  for (int s = 128; s > 0; s >>= 1) {
    if (tid < s) red[tid] = fmaxf(red[tid], red[tid + s]);
    __syncthreads();
  }
  m = red[0]; __syncthreads();
  float sum = 0.f;
#pragma unroll
  for (int j = 0; j < 4; ++j) { v[j] = __expf(v[j] - m); sum += v[j]; }
  red[tid] = sum; __syncthreads();
  for (int s = 128; s > 0; s >>= 1) {
    if (tid < s) red[tid] += red[tid + s];
    __syncthreads();
  }
  const float inv = 1.f / red[0];
#pragma unroll
  for (int j = 0; j < 4; ++j) row[tid + j * 256] = (__bf16)(v[j] * inv);
}

// ---------------- context[b][h][d][e] = (k' . v^T)/N ------------------------

__global__ __launch_bounds__(256) void ctx_gemm(const __bf16* __restrict__ qkv,
                                                float* __restrict__ ctx) {
  __shared__ unsigned int Ks_u[64 * (LD32 / 2)];   // A: 64(d) x 32(n), padded
  __shared__ unsigned int Vs_u[64 * (LD32 / 2)];   // B^T: 64(e) x 32(n), padded
  const __bf16* Ks = (const __bf16*)Ks_u;
  const __bf16* Vs = (const __bf16*)Vs_u;
  const int tid = threadIdx.x, lane = tid & 31, wave = tid >> 5;
  const int mt = wave & 3, nt0 = (wave >> 2) << 1;
  const int h = blockIdx.x, b = blockIdx.y;

  const unsigned int* k_u =
      (const unsigned int*)(qkv + ((size_t)b * O3_ + HID_ + h * 64) * N_);
  const unsigned int* v_u =
      (const unsigned int*)(qkv + ((size_t)b * O3_ + 2 * HID_ + h * 64) * N_);

  v8f acc0 = {}, acc1 = {};
  for (int n0 = 0; n0 < N_; n0 += 32) {
    __syncthreads();
#pragma unroll
    for (int j = 0; j < 4; ++j) {
      int i = tid + j * 256;
      int r = i >> 4, cp = i & 15;
      Ks_u[r * (LD32 / 2) + cp] = k_u[r * (N_ / 2) + (n0 >> 1) + cp];
      Vs_u[r * (LD32 / 2) + cp] = v_u[r * (N_ / 2) + (n0 >> 1) + cp];
    }
    if (n0 + 32 < N_) {
      __builtin_prefetch(&k_u[(tid >> 4) * (N_ / 2) + ((n0 + 32) >> 1) + (tid & 15)], 0, 1);
      __builtin_prefetch(&v_u[(tid >> 4) * (N_ / 2) + ((n0 + 32) >> 1) + (tid & 15)], 0, 1);
    }
    __syncthreads();
    v16bf a  = frag_a(Ks + mt * 16 * LD32, lane, LD32);
    v16bf b0 = frag_bT(Vs + nt0 * 16 * LD32, lane, LD32);
    v16bf b1 = frag_bT(Vs + (nt0 + 1) * 16 * LD32, lane, LD32);
    acc0 = wmma_bf16(a, b0, acc0);
    acc1 = wmma_bf16(a, b1, acc1);
  }
  float* outp = ctx + (size_t)(b * H_ + h) * 64 * 64;
  const int col = lane & 15, rb = (lane >> 4) << 3;
  const float sc = 1.f / (float)N_;   // v/n folded here
#pragma unroll
  for (int r = 0; r < 8; ++r) {
    int row = mt * 16 + rb + r;
    outp[row * 64 + nt0 * 16 + col]       = acc0[r] * sc;
    outp[row * 64 + (nt0 + 1) * 16 + col] = acc1[r] * sc;
  }
}

// ---------------- attn[b][h*64+e][n] = ctx^T[e][d] . q'[d][n] ---------------

__global__ __launch_bounds__(256) void attn_gemm(const float* __restrict__ ctx,
                                                 const __bf16* __restrict__ qkv,
                                                 __bf16* __restrict__ attn) {
  __shared__ unsigned int A_u[64 * (LD64 / 2)];    // A: [e][d] bf16 64x64, padded
  __shared__ unsigned int Bt_u[64 * (LD64 / 2)];   // B^T: [n][d] bf16 64x64, padded
  __bf16* Al = (__bf16*)A_u;
  const __bf16* Ac = (const __bf16*)A_u;
  const __bf16* Bl = (const __bf16*)Bt_u;
  const int tid = threadIdx.x, lane = tid & 31, wave = tid >> 5;
  const int mt = wave & 3, nt0 = (wave >> 2) << 1;
  const int n0 = blockIdx.x * 64;
  const int h = blockIdx.y, b = blockIdx.z;

  // A = ctx^T (f32 -> bf16), transpose on stage; coalesced f32 reads.
  const float* cbase = ctx + (size_t)(b * H_ + h) * 64 * 64;
#pragma unroll
  for (int j = 0; j < 16; ++j) {
    int i = tid + j * 256;
    int d = i >> 6, e = i & 63;
    Al[e * LD64 + d] = (__bf16)cbase[i];
  }
  // B^T = q'^T: 2x2 register transpose, coalesced dword loads.
  const unsigned int* q_u =
      (const unsigned int*)(qkv + ((size_t)b * O3_ + h * 64) * N_);
#pragma unroll
  for (int t = 0; t < 4; ++t) {
    int i = tid + t * 256;           // 0..1023
    int cp = i & 31;                 // n-pair
    int r  = i >> 5;                 // d-pair (0..31)
    unsigned u0 = q_u[(2 * r)     * (N_ / 2) + (n0 >> 1) + cp];
    unsigned u1 = q_u[(2 * r + 1) * (N_ / 2) + (n0 >> 1) + cp];
    Bt_u[(2 * cp)     * (LD64 / 2) + r] = pack_lo(u0, u1);
    Bt_u[(2 * cp + 1) * (LD64 / 2) + r] = pack_hi(u0, u1);
  }
  __syncthreads();
  v8f acc0 = {}, acc1 = {};
#pragma unroll
  for (int kk = 0; kk < 64; kk += 32) {
    v16bf a  = frag_a(Ac + mt * 16 * LD64 + kk, lane, LD64);
    v16bf b0 = frag_bT(Bl + nt0 * 16 * LD64 + kk, lane, LD64);
    v16bf b1 = frag_bT(Bl + (nt0 + 1) * 16 * LD64 + kk, lane, LD64);
    acc0 = wmma_bf16(a, b0, acc0);
    acc1 = wmma_bf16(a, b1, acc1);
  }
  __bf16* outp = attn + ((size_t)b * HID_ + h * 64) * N_;
  const int col = lane & 15, rb = (lane >> 4) << 3;
#pragma unroll
  for (int r = 0; r < 8; ++r) {
    int e = mt * 16 + rb + r;
    outp[(size_t)e * N_ + n0 + nt0 * 16 + col]       = (__bf16)acc0[r];
    outp[(size_t)e * N_ + n0 + (nt0 + 1) * 16 + col] = (__bf16)acc1[r];
  }
}

// ---------------- zero BN stats ---------------------------------------------

__global__ void zero_stats(float* p, int n) {
  int i = blockIdx.x * blockDim.x + threadIdx.x;
  if (i < n) p[i] = 0.f;
}

// ---------------- projection + bias + BN stats accumulation -----------------

__global__ __launch_bounds__(256) void proj_gemm(const __bf16* __restrict__ wo,
                                                 const __bf16* __restrict__ attn,
                                                 const float* __restrict__ bias_out,
                                                 float* __restrict__ out,
                                                 float* __restrict__ gsum,
                                                 float* __restrict__ gsq) {
  __shared__ unsigned int As_u[64 * (LD32 / 2)];
  __shared__ unsigned int Bt_u[64 * (LD32 / 2)];
  __shared__ float csum[64];
  __shared__ float csq[64];
  const __bf16* As = (const __bf16*)As_u;
  const __bf16* Bt = (const __bf16*)Bt_u;
  const int tid = threadIdx.x, lane = tid & 31, wave = tid >> 5;
  const int mt = wave & 3, nt0 = (wave >> 2) << 1;
  const int n0 = blockIdx.x * 64;
  const int o0 = blockIdx.y * 64;
  const int b  = blockIdx.z;
  if (tid < 64) { csum[tid] = 0.f; csq[tid] = 0.f; }

  const unsigned int* wo_u = (const unsigned int*)wo;
  const unsigned int* at_u = (const unsigned int*)(attn + (size_t)b * HID_ * N_);
  v8f acc0 = {}, acc1 = {};
  for (int c0 = 0; c0 < HID_; c0 += 32) {
    __syncthreads();
#pragma unroll
    for (int j = 0; j < 4; ++j) {
      int i = tid + j * 256;
      int r = i >> 4, cp = i & 15;
      As_u[r * (LD32 / 2) + cp] = wo_u[(o0 + r) * (HID_ / 2) + (c0 >> 1) + cp];
    }
#pragma unroll
    for (int t = 0; t < 2; ++t) {
      int i = tid + t * 256;
      int cp = i & 31;
      int r  = i >> 5;
      unsigned u0 = at_u[(c0 + 2 * r)     * (N_ / 2) + (n0 >> 1) + cp];
      unsigned u1 = at_u[(c0 + 2 * r + 1) * (N_ / 2) + (n0 >> 1) + cp];
      Bt_u[(2 * cp)     * (LD32 / 2) + r] = pack_lo(u0, u1);
      Bt_u[(2 * cp + 1) * (LD32 / 2) + r] = pack_hi(u0, u1);
    }
    if (c0 + 32 < HID_) {
      __builtin_prefetch(&wo_u[(o0 + (tid >> 4)) * (HID_ / 2) + ((c0 + 32) >> 1) + (tid & 15)], 0, 1);
      __builtin_prefetch(&at_u[(c0 + 32 + ((tid >> 5) << 1)) * (N_ / 2) + (n0 >> 1) + (tid & 31)], 0, 1);
    }
    __syncthreads();
    v16bf a  = frag_a(As + mt * 16 * LD32, lane, LD32);
    v16bf b0 = frag_bT(Bt + nt0 * 16 * LD32, lane, LD32);
    v16bf b1 = frag_bT(Bt + (nt0 + 1) * 16 * LD32, lane, LD32);
    acc0 = wmma_bf16(a, b0, acc0);
    acc1 = wmma_bf16(a, b1, acc1);
  }
  float* outp = out + (size_t)b * C_ * N_;
  const int col = lane & 15, rb = (lane >> 4) << 3;
#pragma unroll
  for (int r = 0; r < 8; ++r) {
    int lrow = mt * 16 + rb + r;
    int row = o0 + lrow;
    float bi = bias_out[row];
    float v0 = acc0[r] + bi;
    float v1 = acc1[r] + bi;
    outp[(size_t)row * N_ + n0 + nt0 * 16 + col]       = v0;
    outp[(size_t)row * N_ + n0 + (nt0 + 1) * 16 + col] = v1;
    atomicAdd(&csum[lrow], v0 + v1);
    atomicAdd(&csq[lrow], v0 * v0 + v1 * v1);
  }
  __syncthreads();
  if (tid < 64) {
    atomicAdd(&gsum[o0 + tid], csum[tid]);
    atomicAdd(&gsq[o0 + tid], csq[tid]);
  }
}

// ---------------- BN apply (in place on d_out) ------------------------------

__global__ __launch_bounds__(256) void bn_apply(float* __restrict__ out,
                                                const float* __restrict__ gsum,
                                                const float* __restrict__ gsq,
                                                const float* __restrict__ gamma,
                                                const float* __restrict__ beta) {
  const size_t i = (size_t)blockIdx.x * 256 + threadIdx.x;
  const int oc = (int)((i / N_) % C_);
  const float cnt = (float)(B_ * N_);
  float mean = gsum[oc] / cnt;
  float var  = gsq[oc] / cnt - mean * mean;
  float sc = rsqrtf(var + 1e-5f) * gamma[oc];
  float sh = beta[oc] - mean * sc;
  out[i] = out[i] * sc + sh;
}

// ---------------- launch ----------------------------------------------------

extern "C" void kernel_launch(void* const* d_in, const int* in_sizes, int n_in,
                              void* d_out, int out_size, void* d_ws, size_t ws_size,
                              hipStream_t stream) {
  (void)in_sizes; (void)n_in; (void)out_size; (void)ws_size;
  const float* x     = (const float*)d_in[0];
  const float* w_qkv = (const float*)d_in[1];
  const float* w_out = (const float*)d_in[2];
  const float* b_out = (const float*)d_in[3];
  const float* gamma = (const float*)d_in[4];
  const float* beta  = (const float*)d_in[5];
  float* out = (float*)d_out;

  char* ws = (char*)d_ws;
  size_t off = 0;
  auto alloc = [&](size_t bytes) -> void* {
    void* p = ws + off;
    off = (off + bytes + 255) & ~(size_t)255;
    return p;
  };
  __bf16* xb    = (__bf16*)alloc((size_t)B_ * C_ * N_ * 2);
  __bf16* wqb   = (__bf16*)alloc((size_t)O3_ * C_ * 2);
  __bf16* wob   = (__bf16*)alloc((size_t)C_ * HID_ * 2);
  __bf16* qkv   = (__bf16*)alloc((size_t)B_ * O3_ * N_ * 2);
  float*  ctx   = (float*) alloc((size_t)B_ * H_ * 64 * 64 * 4);
  __bf16* attn  = (__bf16*)alloc((size_t)B_ * HID_ * N_ * 2);
  float*  stats = (float*) alloc(2 * C_ * 4);
  float* gsum = stats;
  float* gsq  = stats + C_;

  cvt_f32_bf16<<<(B_ * C_ * N_) / 256, 256, 0, stream>>>(x, xb, B_ * C_ * N_);
  cvt_f32_bf16<<<(O3_ * C_) / 256, 256, 0, stream>>>(w_qkv, wqb, O3_ * C_);
  cvt_f32_bf16<<<(C_ * HID_) / 256, 256, 0, stream>>>(w_out, wob, C_ * HID_);

  gemm_qkv<<<dim3(N_ / 64, O3_ / 64, B_), 256, 0, stream>>>(wqb, xb, qkv);
  softmax_q<<<dim3(N_ / 128, H_, B_), 128, 0, stream>>>(qkv);
  softmax_k<<<dim3(64, H_, B_), 256, 0, stream>>>(qkv);
  ctx_gemm<<<dim3(H_, B_), 256, 0, stream>>>(qkv, ctx);
  attn_gemm<<<dim3(N_ / 64, H_, B_), 256, 0, stream>>>(ctx, qkv, attn);
  zero_stats<<<2, 256, 0, stream>>>(stats, 2 * C_);
  proj_gemm<<<dim3(N_ / 64, C_ / 64, B_), 256, 0, stream>>>(wob, attn, b_out, out,
                                                            gsum, gsq);
  bn_apply<<<(B_ * C_ * N_) / 256, 256, 0, stream>>>(out, gsum, gsq, gamma, beta);
}
